// Transformer_89275190214859
// MI455X (gfx1250) — matrix operand order repaired and checked by reading, not compiled, not run on hardware
//
#include <hip/hip_runtime.h>
#include <hip/hip_bf16.h>

typedef __attribute__((ext_vector_type(16))) __bf16 v16bf;
typedef __attribute__((ext_vector_type(8)))  float  v8f;

#define NTH   256
#define RBLK  64
#define PEDIM 128
#define FDIM  512
#define JT    64
#define KT    32
#define EPSV  1e-5f

#define WMMA_BF16(c, a, b) \
  c = __builtin_amdgcn_wmma_f32_16x16x32_bf16(false, a, false, b, (short)0, c, false, false)

// ---- WMMA fragment loaders (wave32, 16x16x32 bf16 layouts per CDNA5 ISA 7.12.2) ----
// A 16x32 (MxK), row-major source: lanes 0-15 hold M=row, K {0..7,16..23}; lanes 16-31 K {8..15,24..31}
__device__ __forceinline__ v16bf frag_a(const __bf16* base, int ld, int k0, int lane) {
  int row = lane & 15, kh = lane >> 4;
  v16bf a;
#pragma unroll
  for (int i = 0; i < 16; ++i) {
    int k = ((i >> 3) << 4) + (kh << 3) + (i & 7);
    a[i] = base[row * ld + k0 + k];
  }
  return a;
}
// A fragment from transposed storage: element A[m][k] = base[k*ld + m]
__device__ __forceinline__ v16bf frag_aT(const __bf16* base, int ld, int k0, int lane) {
  int row = lane & 15, kh = lane >> 4;
  v16bf a;
#pragma unroll
  for (int i = 0; i < 16; ++i) {
    int k = ((i >> 3) << 4) + (kh << 3) + (i & 7);
    a[i] = base[(k0 + k) * ld + row];
  }
  return a;
}
// B 32x16 (KxN), B[k][n] row-major with row stride ld: lanes 0-15 = cols, K {0..7,16..23}; lanes 16-31 K {8..15,24..31}
__device__ __forceinline__ v16bf frag_b(const __bf16* base, int ld, int k0, int lane) {
  int col = lane & 15, kh = lane >> 4;
  v16bf b;
#pragma unroll
  for (int i = 0; i < 16; ++i) {
    int k = ((i >> 3) << 4) + (kh << 3) + (i & 7);
    b[i] = base[(k0 + k) * ld + col];
  }
  return b;
}
// B fragment where B[k][n] = W[n][k], W row-major with stride ld (i.e. GEMM against W^T)
__device__ __forceinline__ v16bf frag_bT(const __bf16* base, int ld, int k0, int lane) {
  int col = lane & 15, kh = lane >> 4;
  v16bf b;
#pragma unroll
  for (int i = 0; i < 16; ++i) {
    int k = ((i >> 3) << 4) + (kh << 3) + (i & 7);
    b[i] = base[col * ld + k0 + k];
  }
  return b;
}

// out[RBLK][128] = A[RBLK][128] @ W[128][128]^T   (32 tiles, 4 per wave)
__device__ __forceinline__ void gemm_pe128(const __bf16* A, const __bf16* W,
                                           float* outA, int wave, int lane) {
#pragma unroll
  for (int it = 0; it < 4; ++it) {
    int t = wave * 4 + it, tm = t >> 3, tn = t & 7;
    v8f acc = {};
#pragma unroll
    for (int k0 = 0; k0 < PEDIM; k0 += 32) {
      v16bf a = frag_a(A + tm * 16 * PEDIM, PEDIM, k0, lane);
      v16bf b = frag_bT(W + tn * 16 * PEDIM, PEDIM, k0, lane);
      WMMA_BF16(acc, a, b);
    }
    int m0 = tm * 16 + ((lane >> 4) << 3);
    int n  = tn * 16 + (lane & 15);
#pragma unroll
    for (int r = 0; r < 8; ++r) outA[(m0 + r) * PEDIM + n] = acc[r];
  }
}

__global__ void zero_ws_kernel(float* ws, int n) {
  int i = blockIdx.x * blockDim.x + threadIdx.x;
  if (i < n) ws[i] = 0.f;
}

// ---------------- Pass 1: kv = k^T v, ksum = sum(k) ----------------
// shared pool layout (bytes):
//  [0,16K)      kbf   [RBLK][PE] bf16          (persists)
//  [16K,16K+80K) union:
//     phase1: wk 32K | peb 16K | kf 32K
//     phase2: xb 64K | wt 4K   | vbf 8K
//  [96K, +256)  norms[RBLK]
#define SHMEM_A (98304 + 256)
__global__ void kv_reduce_kernel(const float* __restrict__ pe_dst,
                                 const float* __restrict__ x_dst,
                                 const float* __restrict__ Wk,
                                 const float* __restrict__ Wv,
                                 float* __restrict__ kv,
                                 float* __restrict__ ksum,
                                 int N) {
  extern __shared__ char smem[];
  __bf16* kbf = (__bf16*)(smem);
  char* U = smem + 16384;
  __bf16* wk  = (__bf16*)(U);
  __bf16* peb = (__bf16*)(U + 32768);
  float*  kf  = (float*)(U + 49152);
  __bf16* xb  = (__bf16*)(U);
  __bf16* wt  = (__bf16*)(U + 65536);
  __bf16* vbf = (__bf16*)(U + 69632);
  float* norms = (float*)(smem + 98304);

  const int tid = threadIdx.x, wave = tid >> 5, lane = tid & 31;
  const int r0 = blockIdx.x * RBLK;

  for (int i = tid; i < PEDIM * PEDIM; i += NTH) wk[i] = (__bf16)Wk[i];
  for (int i = tid; i < RBLK * PEDIM; i += NTH) {
    int g = r0 + (i >> 7);
    peb[i] = (g < N) ? (__bf16)pe_dst[(size_t)g * PEDIM + (i & 127)] : (__bf16)0.0f;
  }
  __syncthreads();

  // k = pe_dst @ Wk^T
  gemm_pe128(peb, wk, kf, wave, lane);
  __syncthreads();

  if (tid < RBLK) {
    float s = 0.f;
    for (int c = 0; c < PEDIM; ++c) { float v = kf[tid * PEDIM + c]; s += v * v; }
    norms[tid] = (s > 0.f) ? rsqrtf(s) : 0.f;
  }
  __syncthreads();
  for (int i = tid; i < RBLK * PEDIM; i += NTH)
    kbf[i] = (__bf16)(kf[i] * norms[i >> 7]);
  if (tid < PEDIM) {
    float s = 0.f;
    for (int r = 0; r < RBLK; ++r) s += kf[r * PEDIM + tid] * norms[r];
    atomicAdd(&ksum[tid], s);
  }
  __syncthreads();   // kf region about to be overwritten by xb

  for (int i = tid; i < RBLK * FDIM; i += NTH) {
    int g = r0 + (i >> 9);
    xb[i] = (g < N) ? (__bf16)x_dst[(size_t)g * FDIM + (i & 511)] : (__bf16)0.0f;
  }
  __syncthreads();

  for (int j0 = 0; j0 < FDIM; j0 += JT) {
    // v tile [RBLK][JT] = x_dst @ Wv^T (columns j0..j0+63)
    v8f vacc[2];
#pragma unroll
    for (int i = 0; i < 2; ++i) { v8f z = {}; vacc[i] = z; }
    int t0 = wave * 2;
    int tm0 = t0 >> 2, tn0 = t0 & 3, tm1 = (t0 + 1) >> 2, tn1 = (t0 + 1) & 3;
    for (int kf0 = 0; kf0 < FDIM; kf0 += KT) {
      __syncthreads();
      for (int e = tid; e < JT * KT; e += NTH)
        wt[e] = (__bf16)Wv[(size_t)(j0 + (e >> 5)) * FDIM + kf0 + (e & 31)];
      // prefetch next K-step's Wv tile into caches while this one is consumed
      if (kf0 + KT < FDIM)
        __builtin_prefetch(Wv + (size_t)(j0 + (tid & 63)) * FDIM + (kf0 + KT) + ((tid >> 6) << 3), 0, 0);
      __syncthreads();
      {
        v16bf a = frag_a(xb + tm0 * 16 * FDIM, FDIM, kf0, lane);
        v16bf b = frag_bT(wt + tn0 * 16 * KT, KT, 0, lane);
        WMMA_BF16(vacc[0], a, b);
      }
      {
        v16bf a = frag_a(xb + tm1 * 16 * FDIM, FDIM, kf0, lane);
        v16bf b = frag_bT(wt + tn1 * 16 * KT, KT, 0, lane);
        WMMA_BF16(vacc[1], a, b);
      }
    }
    __syncthreads();
#pragma unroll
    for (int it = 0; it < 2; ++it) {
      int t = wave * 2 + it, tm = t >> 2, tn = t & 3;
      int m0 = tm * 16 + ((lane >> 4) << 3);
      int n  = tn * 16 + (lane & 15);
#pragma unroll
      for (int r = 0; r < 8; ++r) vbf[(m0 + r) * JT + n] = (__bf16)vacc[it][r];
    }
    __syncthreads();
    // kv[:, j0:j0+64] += k^T @ v   (out [128][64], 32 tiles, 4 per wave)
#pragma unroll
    for (int it = 0; it < 4; ++it) {
      int t = wave * 4 + it, tp = t >> 2, tj = t & 3;
      v8f acc = {};
#pragma unroll
      for (int rk = 0; rk < RBLK; rk += 32) {
        v16bf a = frag_aT(kbf + tp * 16, PEDIM, rk, lane);
        v16bf b = frag_b(vbf + tj * 16, JT, rk, lane);
        WMMA_BF16(acc, a, b);
      }
      int p0 = tp * 16 + ((lane >> 4) << 3);
      int n  = j0 + tj * 16 + (lane & 15);
#pragma unroll
      for (int r = 0; r < 8; ++r) atomicAdd(&kv[(size_t)(p0 + r) * FDIM + n], acc[r]);
    }
    __syncthreads();
  }
}

// ---------------- Pass 2: per-row output ----------------
// shared pool layout:
//  [0,16K)        qbf  [RBLK][PE] bf16          (persists)
//  [16K,16K+88K)  union:
//     phaseQ: wb 32K | peb 16K | outA 32K
//     phaseM: xb 64K | kvb 16K | wt 4K
//  [104K, +1792) small f32 arrays
#define SHMEM_B (16384 + 90112 + 1792)
__global__ void attn_main_kernel(const float* __restrict__ pe_src,
                                 const float* __restrict__ x_src,
                                 const float* __restrict__ Wq,
                                 const float* __restrict__ Wqs,
                                 const float* __restrict__ Wks,
                                 const float* __restrict__ Wvs,
                                 const float* __restrict__ kv,
                                 const float* __restrict__ ksum,
                                 float* __restrict__ out, int N) {
  extern __shared__ char smem[];
  __bf16* qbf = (__bf16*)smem;
  char* U = smem + 16384;
  __bf16* wb   = (__bf16*)(U);
  __bf16* peb  = (__bf16*)(U + 32768);
  float*  outA = (float*)(U + 49152);
  __bf16* xb   = (__bf16*)(U);
  __bf16* kvb  = (__bf16*)(U + 65536);
  __bf16* wt   = (__bf16*)(U + 81920);
  float* sm    = (float*)(smem + 16384 + 90112);
  float* score = sm;          // [64]
  float* invdn = sm + 64;     // [64]
  float* qdot  = sm + 128;    // [64]
  float* ssU   = sm + 192;    // [64]
  float* dotUW = sm + 256;    // [64]
  float* ssW   = sm + 320;    // [64]
  float* ntmp  = sm + 384;    // [64]

  const int tid = threadIdx.x, wave = tid >> 5, lane = tid & 31;
  const int r0 = blockIdx.x * RBLK;

  for (int i = tid; i < RBLK * PEDIM; i += NTH) {
    int g = r0 + (i >> 7);
    peb[i] = (g < N) ? (__bf16)pe_src[(size_t)g * PEDIM + (i & 127)] : (__bf16)0.0f;
  }
  if (tid < RBLK) { dotUW[tid] = 0.f; ssW[tid] = 0.f; }
  __syncthreads();

  // ---- q = l2norm(pe_src @ Wq^T) ----
  for (int i = tid; i < PEDIM * PEDIM; i += NTH) wb[i] = (__bf16)Wq[i];
  __syncthreads();
  gemm_pe128(peb, wb, outA, wave, lane);
  __syncthreads();
  if (tid < RBLK) {
    float s = 0.f;
    for (int c = 0; c < PEDIM; ++c) { float v = outA[tid * PEDIM + c]; s += v * v; }
    ntmp[tid] = (s > 0.f) ? rsqrtf(s) : 0.f;
  }
  __syncthreads();
  for (int i = tid; i < RBLK * PEDIM; i += NTH)
    qbf[i] = (__bf16)(outA[i] * ntmp[i >> 7]);
  if (tid < RBLK) {
    float inv = ntmp[tid], s = 0.f;
    for (int c = 0; c < PEDIM; ++c) s += outA[tid * PEDIM + c] * inv * ksum[c];
    qdot[tid] = s;                    // q_hat . ksum
  }
  __syncthreads();

  // ---- u = pe_src @ Wqs^T ----
  for (int i = tid; i < PEDIM * PEDIM; i += NTH) wb[i] = (__bf16)Wqs[i];
  __syncthreads();
  gemm_pe128(peb, wb, outA, wave, lane);
  __syncthreads();
  if (tid < RBLK) {
    float s = 0.f;
    for (int c = 0; c < PEDIM; ++c) { float v = outA[tid * PEDIM + c]; s += v * v; }
    ssU[tid] = s;
  }
  __syncthreads();

  // ---- w = pe_src @ Wks^T, fused row-dot with u (never materialized) ----
  for (int i = tid; i < PEDIM * PEDIM; i += NTH) wb[i] = (__bf16)Wks[i];
  __syncthreads();
#pragma unroll
  for (int it = 0; it < 4; ++it) {
    int t = wave * 4 + it, tm = t >> 3, tn = t & 7;
    v8f acc = {};
#pragma unroll
    for (int k0 = 0; k0 < PEDIM; k0 += 32) {
      v16bf a = frag_a(peb + tm * 16 * PEDIM, PEDIM, k0, lane);
      v16bf b = frag_bT(wb + tn * 16 * PEDIM, PEDIM, k0, lane);
      WMMA_BF16(acc, a, b);
    }
    int m0 = tm * 16 + ((lane >> 4) << 3);
    int n  = tn * 16 + (lane & 15);
#pragma unroll
    for (int r = 0; r < 8; ++r) {
      float w = acc[r];
      atomicAdd(&dotUW[m0 + r], w * outA[(m0 + r) * PEDIM + n]);
      atomicAdd(&ssW[m0 + r], w * w);
    }
  }
  __syncthreads();
  if (tid < RBLK) {
    float denom = sqrtf(ssU[tid] * ssW[tid]);
    float sc = (denom > 0.f) ? dotUW[tid] / denom : 0.f;   // q_self . k_self (normalized)
    score[tid] = sc;
    invdn[tid] = 1.f / (qdot[tid] + sc + EPSV);
  }
  __syncthreads();

  // ---- stage x_src rows (overwrites phaseQ union region) ----
  for (int i = tid; i < RBLK * FDIM; i += NTH) {
    int g = r0 + (i >> 9);
    xb[i] = (g < N) ? (__bf16)x_src[(size_t)g * FDIM + (i & 511)] : (__bf16)0.0f;
  }
  __syncthreads();

  for (int j0 = 0; j0 < FDIM; j0 += JT) {
    for (int e = tid; e < PEDIM * JT; e += NTH)
      kvb[e] = (__bf16)kv[(size_t)(e >> 6) * FDIM + j0 + (e & 63)];
    // prefetch next j-tile of kv
    if (j0 + JT < FDIM)
      __builtin_prefetch(kv + (size_t)(tid >> 1) * FDIM + (j0 + JT) + ((tid & 1) << 5), 0, 0);
    __syncthreads();

    int t0 = wave * 2;
    int tm0 = t0 >> 2, tn0 = t0 & 3, tm1 = (t0 + 1) >> 2, tn1 = (t0 + 1) & 3;
    v8f up0 = {}, up1 = {}, vs0 = {}, vs1 = {};

    // up = q_hat @ kv[:, tile]
#pragma unroll
    for (int k0 = 0; k0 < PEDIM; k0 += 32) {
      v16bf a0 = frag_a(qbf + tm0 * 16 * PEDIM, PEDIM, k0, lane);
      v16bf b0 = frag_b(kvb + tn0 * 16, JT, k0, lane);
      WMMA_BF16(up0, a0, b0);
      v16bf a1 = frag_a(qbf + tm1 * 16 * PEDIM, PEDIM, k0, lane);
      v16bf b1 = frag_b(kvb + tn1 * 16, JT, k0, lane);
      WMMA_BF16(up1, a1, b1);
    }
    // v_self tile = x_src @ Wvs^T
    for (int kf0 = 0; kf0 < FDIM; kf0 += KT) {
      __syncthreads();
      for (int e = tid; e < JT * KT; e += NTH)
        wt[e] = (__bf16)Wvs[(size_t)(j0 + (e >> 5)) * FDIM + kf0 + (e & 31)];
      // prefetch next K-step's Wvs tile
      if (kf0 + KT < FDIM)
        __builtin_prefetch(Wvs + (size_t)(j0 + (tid & 63)) * FDIM + (kf0 + KT) + ((tid >> 6) << 3), 0, 0);
      __syncthreads();
      {
        v16bf a = frag_a(xb + tm0 * 16 * FDIM, FDIM, kf0, lane);
        v16bf b = frag_bT(wt + tn0 * 16 * KT, KT, 0, lane);
        WMMA_BF16(vs0, a, b);
      }
      {
        v16bf a = frag_a(xb + tm1 * 16 * FDIM, FDIM, kf0, lane);
        v16bf b = frag_bT(wt + tn1 * 16 * KT, KT, 0, lane);
        WMMA_BF16(vs1, a, b);
      }
    }
    // combine + store
    {
      int m0 = tm0 * 16 + ((lane >> 4) << 3), n0 = tn0 * 16 + (lane & 15);
#pragma unroll
      for (int r = 0; r < 8; ++r) {
        int m = m0 + r, g = r0 + m;
        if (g < N) out[(size_t)g * FDIM + j0 + n0] = (up0[r] + score[m] * vs0[r]) * invdn[m];
      }
      int m1 = tm1 * 16 + ((lane >> 4) << 3), n1 = tn1 * 16 + (lane & 15);
#pragma unroll
      for (int r = 0; r < 8; ++r) {
        int m = m1 + r, g = r0 + m;
        if (g < N) out[(size_t)g * FDIM + j0 + n1] = (up1[r] + score[m] * vs1[r]) * invdn[m];
      }
    }
    __syncthreads();   // protect kvb before restage
  }
}

extern "C" void kernel_launch(void* const* d_in, const int* in_sizes, int n_in,
                              void* d_out, int out_size, void* d_ws, size_t ws_size,
                              hipStream_t stream) {
  (void)n_in; (void)out_size; (void)ws_size;
  const float* pe_src = (const float*)d_in[0];
  const float* pe_dst = (const float*)d_in[1];
  const float* x_src  = (const float*)d_in[2];
  const float* x_dst  = (const float*)d_in[3];
  const float* Wq     = (const float*)d_in[4];
  const float* Wk     = (const float*)d_in[5];
  const float* Wv     = (const float*)d_in[6];
  const float* Wqs    = (const float*)d_in[7];
  const float* Wks    = (const float*)d_in[8];
  const float* Wvs    = (const float*)d_in[9];
  float* out = (float*)d_out;

  const int N = in_sizes[0] / PEDIM;
  float* kv   = (float*)d_ws;            // [PE][F]
  float* ksum = kv + PEDIM * FDIM;       // [PE]

  const int nz = PEDIM * FDIM + PEDIM;
  zero_ws_kernel<<<(nz + 255) / 256, 256, 0, stream>>>(kv, nz);

  const int blocks = (N + RBLK - 1) / RBLK;
  kv_reduce_kernel<<<blocks, NTH, SHMEM_A, stream>>>(pe_dst, x_dst, Wk, Wv, kv, ksum, N);
  attn_main_kernel<<<blocks, NTH, SHMEM_B, stream>>>(pe_src, x_src, Wq, Wqs, Wks, Wvs,
                                                     kv, ksum, out, N);
}
